// MiniTriangularUpdate_317827580831
// MI455X (gfx1250) — compile-verified, matrix operand order checked by reading, not run on hardware
//
#include <hip/hip_runtime.h>

// MI455X / gfx1250 triangle multiplicative update.
// All matmuls via v_wmma_f32_16x16x32_bf16 (bf16 operands, fp32 accumulate).
// Triangle GEMM stages tiles through double-buffered LDS using the CDNA5
// async-to-LDS path (ASYNCcnt) when the toolchain exposes it.
// Workspace requirement: ~227 MB.

typedef __attribute__((ext_vector_type(16))) __bf16 v16bf;
typedef __attribute__((ext_vector_type(8)))  float  v8f;
typedef __attribute__((ext_vector_type(4)))  int    v4i;

#define NN 768
#define DD 128
#define DH 64
#define LN_EPS 1e-5f

#if defined(__has_builtin)
#if __has_builtin(__builtin_amdgcn_global_load_async_to_lds_b128) && \
    __has_builtin(__builtin_amdgcn_s_wait_asynccnt)
#define USE_ASYNC_LDS 1
#endif
#endif

static __device__ __forceinline__ v8f wmma_bf16(v16bf a, v16bf b, v8f c) {
  // 8 args: (neg_a, A, neg_b, B, c_mod, C, reuse_a, reuse_b)
  return __builtin_amdgcn_wmma_f32_16x16x32_bf16(false, a, false, b,
                                                 (short)0, c, false, false);
}

static __device__ __forceinline__ float fast_rcp(float x) {
#if defined(__has_builtin)
#if __has_builtin(__builtin_amdgcn_rcpf)
  return __builtin_amdgcn_rcpf(x);
#else
  return 1.0f / x;
#endif
#else
  return 1.0f / x;
#endif
}

static __device__ __forceinline__ float sigmoidf_(float x) {
  return fast_rcp(1.0f + __expf(-x));   // v_exp_f32 + v_rcp_f32
}

// Copy 16B global -> LDS, async when available.
static __device__ __forceinline__ void cp16(const __bf16* g, __bf16* l) {
#ifdef USE_ASYNC_LDS
  __builtin_amdgcn_global_load_async_to_lds_b128(
      (__attribute__((address_space(1))) v4i*)g,
      (__attribute__((address_space(3))) v4i*)l, 0, 0);
#else
  *(int4*)l = *(const int4*)g;
#endif
}

#ifdef USE_ASYNC_LDS
#define ASYNC_WAIT(n) __builtin_amdgcn_s_wait_asynccnt(n)
#else
#define ASYNC_WAIT(n)
#endif

// ---------------------------------------------------------------------------
// Weight fp32 -> bf16 conversion into workspace.
// dst layout (elements): [0,16384) Wp_in, [16384,32768) Wg_in,
//                        [32768,40960) Wp_out, [40960,49152) Wg_out
// ---------------------------------------------------------------------------
__global__ void k_prep_weights(const float* __restrict__ wpi,
                               const float* __restrict__ wgi,
                               const float* __restrict__ wpo,
                               const float* __restrict__ wgo,
                               __bf16* __restrict__ dst) {
  int i = blockIdx.x * blockDim.x + threadIdx.x;
  if (i < 16384)      dst[i] = (__bf16)wpi[i];
  else if (i < 32768) dst[i] = (__bf16)wgi[i - 16384];
  else if (i < 40960) dst[i] = (__bf16)wpo[i - 32768];
  else if (i < 49152) dst[i] = (__bf16)wgo[i - 40960];
}

// ---------------------------------------------------------------------------
// Kernel 1: LayerNorm(D=128) + gated input projection + mask, then scatter the
// four 32-channel quarters into d-major [32][768][768] bf16 layouts so that
// both triangle einsums become plain C = A * B^T GEMMs.
//   a1L[d][i][k] = h[i,k, 0+d]     b1L[d][j][k] = h[j,k,32+d]
//   a2L[d][i][k] = h[k,i,64+d]     b2L[d][j][k] = h[k,j,96+d]   (transposed)
// Block: 256 threads (8 waves) -> 32 rows of the flattened (768*768, 128) mat.
// ---------------------------------------------------------------------------
#define SWS 144   // LDS K-stride for transposed 128x128 weights (288B = 9*32)
#define SHS 144   // LDS row stride for normalized activations

__global__ __launch_bounds__(256) void k_fused_in(
    const float* __restrict__ x, const float* __restrict__ mask,
    const float* __restrict__ wn, const float* __restrict__ bn,
    const __bf16* __restrict__ Wp, const __bf16* __restrict__ Wg,
    __bf16* __restrict__ a1, __bf16* __restrict__ b1,
    __bf16* __restrict__ a2, __bf16* __restrict__ b2) {
  __shared__ __attribute__((aligned(32))) __bf16 sW[2][DD * SWS];
  __shared__ __attribute__((aligned(32))) __bf16 sH[32 * SHS];
  __shared__ float sMask[32];

  const int tid = threadIdx.x, lane = tid & 31, wave = tid >> 5;
  const long r0 = (long)blockIdx.x * 32;
  const int p = (int)(r0 / NN), q0 = (int)(r0 % NN);

  // Stage both weight matrices transposed into LDS (vector global reads,
  // scattered LDS writes).
  for (int m = 0; m < 2; ++m) {
    const __bf16* src = m ? Wg : Wp;
    for (int ch = tid; ch < (DD * DD) / 8; ch += 256) {
      const int k = ch >> 4, c0 = (ch & 15) * 8;   // flat = k*128 + c
      union { int4 q; __bf16 h[8]; } u;
      u.q = *(const int4*)(src + (size_t)k * DD + c0);
#pragma unroll
      for (int t = 0; t < 8; ++t) sW[m][(c0 + t) * SWS + k] = u.h[t];
    }
  }
  if (tid < 32) sMask[tid] = mask[(long)p * NN + q0 + tid];

  // LayerNorm: each wave normalizes 4 rows; 4 channels per lane.
#pragma unroll
  for (int rr = 0; rr < 4; ++rr) {
    const int row = wave * 4 + rr;
    const float* xr = x + (r0 + row) * (long)DD;
    float v[4];
#pragma unroll
    for (int t = 0; t < 4; ++t) v[t] = xr[lane + 32 * t];
    float s = v[0] + v[1] + v[2] + v[3];
#pragma unroll
    for (int o = 16; o >= 1; o >>= 1) s += __shfl_xor(s, o, 32);
    const float mean = s * (1.0f / 128.0f);
    float ss = 0.f;
#pragma unroll
    for (int t = 0; t < 4; ++t) { float d = v[t] - mean; ss += d * d; }
#pragma unroll
    for (int o = 16; o >= 1; o >>= 1) ss += __shfl_xor(ss, o, 32);
    const float rstd = rsqrtf(ss * (1.0f / 128.0f) + LN_EPS);
#pragma unroll
    for (int t = 0; t < 4; ++t) {
      const int c = lane + 32 * t;
      sH[row * SHS + c] = (__bf16)((v[t] - mean) * rstd * wn[c] + bn[c]);
    }
  }
  __syncthreads();

  // GEMM: wave -> row-half rh (16 rows) x two 16-col tiles, P and G. K=128.
  const int lr = lane & 15, lh = lane >> 4;
  const int rh = wave >> 2;             // 0 or 1: rows rh*16 .. rh*16+15
#pragma unroll
  for (int t = 0; t < 2; ++t) {
    const int col0 = ((wave & 3) * 2 + t) * 16;
    v8f accP = {}; v8f accG = {};
#pragma unroll
    for (int ks = 0; ks < 4; ++ks) {
      const v16bf af = *(const v16bf*)&sH[(rh * 16 + lr) * SHS + ks * 32 + lh * 16];
      const v16bf bP = *(const v16bf*)&sW[0][(col0 + lr) * SWS + ks * 32 + lh * 16];
      const v16bf bG = *(const v16bf*)&sW[1][(col0 + lr) * SWS + ks * 32 + lh * 16];
      accP = wmma_bf16(af, bP, accP);
      accG = wmma_bf16(af, bG, accG);
    }
    // Gate + mask + quarter scatter. C/D layout: VGPR e, lanes0-15 -> M=e,
    // lanes16-31 -> M=e+8; N = lane&15.
#pragma unroll
    for (int e = 0; e < 8; ++e) {
      const int M = rh * 16 + e + 8 * lh;
      const int c = col0 + lr;
      const float hv = accP[e] * sigmoidf_(accG[e]) * sMask[M];
      const __bf16 hb = (__bf16)hv;
      const int d = c & 31, quarter = c >> 5;
      const long q = q0 + M;
      if (quarter == 0)      a1[((long)d * NN + p) * NN + q] = hb;
      else if (quarter == 1) b1[((long)d * NN + p) * NN + q] = hb;
      else if (quarter == 2) a2[((long)d * NN + q) * NN + p] = hb;
      else                   b2[((long)d * NN + q) * NN + p] = hb;
    }
  }
}

// ---------------------------------------------------------------------------
// Kernel 2: triangle GEMM, one of 32 d-slices per grid.z:
//   C[i][j] = sum_k A[i][k] * B[j][k]   (A*B^T, both row-major bf16 768x768)
// Block = 256 threads (8 waves, 4x2) -> 128x64 output tile. K staged through
// double-buffered LDS (async-to-LDS when available). Each wave: 32x32 tile,
// 4 WMMA per 32-wide K step, 96 v_wmma total.
// ---------------------------------------------------------------------------
#define STS 48    // LDS K-stride for staged tiles (96B = 3*32)

__global__ __launch_bounds__(256) void k_triangle(
    const __bf16* __restrict__ Aq, const __bf16* __restrict__ Bq,
    __bf16* __restrict__ outT, int halfsel) {
  __shared__ __attribute__((aligned(32))) __bf16 sA[2][128 * STS];
  __shared__ __attribute__((aligned(32))) __bf16 sB[2][64 * STS];

  const int tid = threadIdx.x, lane = tid & 31, wave = tid >> 5;
  const int d = blockIdx.z;
  const int i0b = blockIdx.x * 128, j0b = blockIdx.y * 64;
  const int wi = wave >> 1, wj = wave & 1;            // 4x2 wave grid
  const int i0 = wi * 32, j0 = wj * 32;               // within block tile
  const __bf16* Am = Aq + (size_t)d * NN * NN;
  const __bf16* Bm = Bq + (size_t)d * NN * NN;

  // Per-thread staging assignments (16B chunks): A = 128x32 (512 chunks, 2
  // per thread), B = 64x32 (256 chunks, 1 per thread).
  const int rA0 = tid >> 2, kA0 = (tid & 3) * 8;
  const int rA1 = (tid + 256) >> 2, kA1 = ((tid + 256) & 3) * 8;

  auto stage = [&](int buf, int k0) {
    cp16(&Am[(size_t)(i0b + rA0) * NN + k0 + kA0], &sA[buf][rA0 * STS + kA0]);
    cp16(&Am[(size_t)(i0b + rA1) * NN + k0 + kA1], &sA[buf][rA1 * STS + kA1]);
    cp16(&Bm[(size_t)(j0b + rA0) * NN + k0 + kA0], &sB[buf][rA0 * STS + kA0]);
  };

  const int lr = lane & 15, lh = lane >> 4;
  v8f acc00 = {}, acc01 = {}, acc10 = {}, acc11 = {};

  stage(0, 0);
  for (int ks = 0; ks < NN / 32; ++ks) {
    const int cur = ks & 1;
    if (ks + 1 < NN / 32) {
      stage(1 - cur, (ks + 1) * 32);
      ASYNC_WAIT(3);        // 3 newer ops outstanding -> stage `cur` landed
    } else {
      ASYNC_WAIT(0);
    }
    __syncthreads();

    const v16bf a0 = *(const v16bf*)&sA[cur][(i0 + lr) * STS + lh * 16];
    const v16bf a1 = *(const v16bf*)&sA[cur][(i0 + 16 + lr) * STS + lh * 16];
    const v16bf b0 = *(const v16bf*)&sB[cur][(j0 + lr) * STS + lh * 16];
    const v16bf b1 = *(const v16bf*)&sB[cur][(j0 + 16 + lr) * STS + lh * 16];
    acc00 = wmma_bf16(a0, b0, acc00);
    acc01 = wmma_bf16(a0, b1, acc01);
    acc10 = wmma_bf16(a1, b0, acc10);
    acc11 = wmma_bf16(a1, b1, acc11);
    __syncthreads();
  }

  const int colbase = halfsel * 32 + d;
  const int gi = i0b + i0, gj = j0b + j0;
#pragma unroll
  for (int e = 0; e < 8; ++e) {
    const int M = e + 8 * lh, Nc = lr;
    outT[((size_t)(gi + M) * NN + (gj + Nc)) * DH + colbase]           = (__bf16)acc00[e];
    outT[((size_t)(gi + M) * NN + (gj + 16 + Nc)) * DH + colbase]      = (__bf16)acc01[e];
    outT[((size_t)(gi + 16 + M) * NN + (gj + Nc)) * DH + colbase]      = (__bf16)acc10[e];
    outT[((size_t)(gi + 16 + M) * NN + (gj + 16 + Nc)) * DH + colbase] = (__bf16)acc11[e];
  }
}

// ---------------------------------------------------------------------------
// Kernel 3: LayerNorm(64) + gated output projection (K=64 -> N=128), fp32 out.
// Block: 256 threads (8 waves) -> 32 rows.
// ---------------------------------------------------------------------------
#define SWO 80   // 160B stride = 5*32 -> aligned 32B fragment loads
#define SHO 80

__global__ __launch_bounds__(256) void k_fused_out(
    const __bf16* __restrict__ tIn,
    const float* __restrict__ wn, const float* __restrict__ bn,
    const __bf16* __restrict__ Wp, const __bf16* __restrict__ Wg,
    float* __restrict__ out) {
  __shared__ __attribute__((aligned(32))) __bf16 sW[2][DD * SWO];
  __shared__ __attribute__((aligned(32))) __bf16 sH[32 * SHO];
  const int tid = threadIdx.x, lane = tid & 31, wave = tid >> 5;
  const long r0 = (long)blockIdx.x * 32;

  for (int m = 0; m < 2; ++m) {
    const __bf16* src = m ? Wg : Wp;
    for (int ch = tid; ch < (DH * DD) / 8; ch += 256) {
      const int k = ch >> 4, c0 = (ch & 15) * 8;   // flat = k*128 + c, k<64
      union { int4 q; __bf16 h[8]; } u;
      u.q = *(const int4*)(src + (size_t)k * DD + c0);
#pragma unroll
      for (int t = 0; t < 8; ++t) sW[m][(c0 + t) * SWO + k] = u.h[t];
    }
  }

  // LayerNorm over 64 channels; 2 contiguous channels per lane (32b loads).
#pragma unroll
  for (int rr = 0; rr < 4; ++rr) {
    const int row = wave * 4 + rr;
    const __bf16* tr = tIn + (r0 + row) * (long)DH;
    union { unsigned u; __bf16 h[2]; } pr;
    pr.u = *(const unsigned*)(tr + 2 * lane);
    const float v0 = (float)pr.h[0], v1 = (float)pr.h[1];
    float s = v0 + v1;
#pragma unroll
    for (int o = 16; o >= 1; o >>= 1) s += __shfl_xor(s, o, 32);
    const float mean = s * (1.0f / 64.0f);
    const float d0 = v0 - mean, d1 = v1 - mean;
    float ss = d0 * d0 + d1 * d1;
#pragma unroll
    for (int o = 16; o >= 1; o >>= 1) ss += __shfl_xor(ss, o, 32);
    const float rstd = rsqrtf(ss * (1.0f / 64.0f) + LN_EPS);
    const int c0 = 2 * lane;
    sH[row * SHO + c0]     = (__bf16)(d0 * rstd * wn[c0] + bn[c0]);
    sH[row * SHO + c0 + 1] = (__bf16)(d1 * rstd * wn[c0 + 1] + bn[c0 + 1]);
  }
  __syncthreads();

  const int lr = lane & 15, lh = lane >> 4;
  const int rh = wave >> 2;
#pragma unroll
  for (int t = 0; t < 2; ++t) {
    const int col0 = ((wave & 3) * 2 + t) * 16;
    v8f accP = {}; v8f accG = {};
#pragma unroll
    for (int ks = 0; ks < 2; ++ks) {   // K = 64
      const v16bf af = *(const v16bf*)&sH[(rh * 16 + lr) * SHO + ks * 32 + lh * 16];
      const v16bf bP = *(const v16bf*)&sW[0][(col0 + lr) * SWO + ks * 32 + lh * 16];
      const v16bf bG = *(const v16bf*)&sW[1][(col0 + lr) * SWO + ks * 32 + lh * 16];
      accP = wmma_bf16(af, bP, accP);
      accG = wmma_bf16(af, bG, accG);
    }
#pragma unroll
    for (int e = 0; e < 8; ++e) {
      const int M = rh * 16 + e + 8 * lh;
      out[(r0 + M) * (long)DD + col0 + lr] = accP[e] * sigmoidf_(accG[e]);
    }
  }
}

// ---------------------------------------------------------------------------
extern "C" void kernel_launch(void* const* d_in, const int* in_sizes, int n_in,
                              void* d_out, int out_size, void* d_ws, size_t ws_size,
                              hipStream_t stream) {
  const float* x    = (const float*)d_in[0];
  const float* mask = (const float*)d_in[1];
  const float* wni  = (const float*)d_in[2];
  const float* bni  = (const float*)d_in[3];
  const float* Wpi  = (const float*)d_in[4];
  const float* Wgi  = (const float*)d_in[5];
  const float* wno  = (const float*)d_in[6];
  const float* bno  = (const float*)d_in[7];
  const float* Wpo  = (const float*)d_in[8];
  const float* Wgo  = (const float*)d_in[9];
  float* out = (float*)d_out;

  char* ws = (char*)d_ws;
  __bf16* wts = (__bf16*)ws;                                   // 49152 elems
  const size_t QOFF   = 131072;
  const size_t QBYTES = (size_t)32 * NN * NN * sizeof(__bf16); // 37,748,736 B
  __bf16* a1 = (__bf16*)(ws + QOFF);
  __bf16* b1 = (__bf16*)(ws + QOFF + 1 * QBYTES);
  __bf16* a2 = (__bf16*)(ws + QOFF + 2 * QBYTES);
  __bf16* b2 = (__bf16*)(ws + QOFF + 3 * QBYTES);
  __bf16* tb = (__bf16*)(ws + QOFF + 4 * QBYTES);              // 768*768*64 bf16

  k_prep_weights<<<192, 256, 0, stream>>>(Wpi, Wgi, Wpo, Wgo, wts);

  k_fused_in<<<(NN * NN) / 32, 256, 0, stream>>>(
      x, mask, wni, bni, wts, wts + 16384, a1, b1, a2, b2);

  dim3 gtri(NN / 128, NN / 64, 32);
  k_triangle<<<gtri, 256, 0, stream>>>(a1, b1, tb, 0);
  k_triangle<<<gtri, 256, 0, stream>>>(a2, b2, tb, 1);

  k_fused_out<<<(NN * NN) / 32, 256, 0, stream>>>(
      tb, wno, bno, wts + 32768, wts + 40960, out);
}
